// SelfAttentionLayer_34127810134762
// MI455X (gfx1250) — compile-verified, matrix operand order checked by reading, not run on hardware
//
#include <hip/hip_runtime.h>

// ---------------------------------------------------------------------------
// Self-attention (B=4, N=4096, D=256, fp32 ref) for gfx1250 / MI455X.
//  - bf16 WMMA (v_wmma_f32_16x16x32_bf16), fp32 accumulation
//  - flash attention (online softmax), 64-key blocks
//  - Q pre-scaled by (1/sqrt(D))*log2(e) at projection -> softmax uses raw
//    v_exp_f32 (exp2) with no per-element multiplies
//  - V written pre-transposed [b][d][n] by the projection kernel so both K
//    and V staging are straight copies done with global_load_async_to_lds_b128
//    (ASYNCcnt), double-buffered in LDS to overlap with WMMA compute
// ---------------------------------------------------------------------------

typedef __attribute__((ext_vector_type(16))) __bf16 v16bf;
typedef __attribute__((ext_vector_type(8)))  float  v8f;

union BF16x16 {
  v16bf v;
  uint4 q[2];
  unsigned short u[16];
};

__device__ __forceinline__ unsigned short f2bf(float f) {
  union { __bf16 b; unsigned short u; } c;
  c.b = (__bf16)f;  // native cvt, RNE
  return c.u;
}

__device__ __forceinline__ v8f vzero8() {
  v8f v;
#pragma unroll
  for (int i = 0; i < 8; ++i) v[i] = 0.0f;
  return v;
}

// Load 16 bf16 in WMMA A/B per-lane order: 8 at base, 8 at base+16 elements.
__device__ __forceinline__ v16bf load16(const unsigned short* base) {
  BF16x16 t;
  t.q[0] = *(const uint4*)(base);
  t.q[1] = *(const uint4*)(base + 16);
  return t.v;
}

__device__ __forceinline__ v16bf cvt16(const float* p) {
  BF16x16 t;
#pragma unroll
  for (int i = 0; i < 8; ++i) t.u[i] = f2bf(p[i]);
#pragma unroll
  for (int i = 0; i < 8; ++i) t.u[8 + i] = f2bf(p[16 + i]);
  return t.v;
}

__device__ __forceinline__ v8f wmma_bf16(v16bf a, v16bf b, v8f c) {
  return __builtin_amdgcn_wmma_f32_16x16x32_bf16(false, a, false, b,
                                                 (short)0, c, false, false);
}

// Async 16B copy global -> LDS (per lane).  LDS offset = low 32 bits of the
// generic shared-aperture address.  Tracked by ASYNCcnt.
__device__ __forceinline__ void async_copy_b128(void* lds, const void* g) {
  unsigned int ldsoff = (unsigned int)(unsigned long long)lds;
  asm volatile("global_load_async_to_lds_b128 %0, %1, off"
               :
               : "v"(ldsoff), "v"(g)
               : "memory");
}

__device__ __forceinline__ void wait_async0() {
  asm volatile("s_wait_asynccnt 0" ::: "memory");
}

#define SCALE_LOG2E 0.09016994f  /* (1/16) * log2(e) */

// ---------------------------------------------------------------------------
// Kernel 1: QKV projection. grid=(256,3) z: 0=Q (pre-scaled), 1=K, 2=V
// (transposed [b][d][n]).  block=128 (4 waves), wave -> 16x256 output block.
// ---------------------------------------------------------------------------
__global__ __launch_bounds__(128) void qkv_proj_kernel(
    const float* __restrict__ x,
    const float* __restrict__ Wq, const float* __restrict__ bq,
    const float* __restrict__ Wk, const float* __restrict__ bk,
    const float* __restrict__ Wv, const float* __restrict__ bv,
    unsigned short* __restrict__ Qw, unsigned short* __restrict__ Kw,
    unsigned short* __restrict__ Vtw) {
  __shared__ unsigned short Wt[256 * 32];  // [n][k] bf16 for current k-block

  const int tid = threadIdx.x;
  const int lane = tid & 31;
  const int wave = tid >> 5;
  const int l16 = lane & 15;
  const int half = lane >> 4;

  const int z = blockIdx.y;
  const float* W = (z == 0) ? Wq : (z == 1) ? Wk : Wv;
  const float* bias = (z == 0) ? bq : (z == 1) ? bk : bv;

  const int m0 = blockIdx.x * 64 + wave * 16;
  const int arow = m0 + l16;

  v8f acc[16];
#pragma unroll
  for (int nt = 0; nt < 16; ++nt) acc[nt] = vzero8();

  for (int kblk = 0; kblk < 8; ++kblk) {
    __syncthreads();
    for (int i = tid; i < 32 * 256; i += 128) {
      const int k = i >> 8;
      const int n = i & 255;
      Wt[n * 32 + k] = f2bf(W[(kblk * 32 + k) * 256 + n]);
    }
    __syncthreads();

    const v16bf a = cvt16(&x[arow * 256 + kblk * 32 + half * 8]);
#pragma unroll
    for (int nt = 0; nt < 16; ++nt) {
      const v16bf bmat = load16(&Wt[(nt * 16 + l16) * 32 + half * 8]);
      acc[nt] = wmma_bf16(a, bmat, acc[nt]);
    }
  }

  // Epilogue: bias (+ Q scale fold), store bf16.
  if (z == 2) {
    // V transposed: Vtw[((b*256 + col)*4096) + n]
    const int bb = m0 >> 12;
    const int n0 = m0 & 4095;
#pragma unroll
    for (int nt = 0; nt < 16; ++nt) {
      const float bvl = bias[nt * 16 + l16];
      unsigned short* col = &Vtw[((size_t)bb * 256 + nt * 16 + l16) * 4096 + n0];
#pragma unroll
      for (int r = 0; r < 8; ++r)
        col[r + 8 * half] = f2bf(acc[nt][r] + bvl);
    }
  } else {
    const float mulf = (z == 0) ? SCALE_LOG2E : 1.0f;
    unsigned short* dst = (z == 0) ? Qw : Kw;
#pragma unroll
    for (int nt = 0; nt < 16; ++nt) {
      const float bvl = bias[nt * 16 + l16];
#pragma unroll
      for (int r = 0; r < 8; ++r)
        dst[(m0 + r + 8 * half) * 256 + nt * 16 + l16] =
            f2bf((acc[nt][r] + bvl) * mulf);
    }
  }
}

// ---------------------------------------------------------------------------
// Kernel 2: flash attention, 64-key blocks, double-buffered async staging.
// grid=(64,4), block=128 (4 waves); wave owns 16 query rows.
// ---------------------------------------------------------------------------
__global__ __launch_bounds__(128) void flash_attn_kernel(
    const unsigned short* __restrict__ Qw,
    const unsigned short* __restrict__ Kw,
    const unsigned short* __restrict__ Vtw,
    float* __restrict__ out) {
  __shared__ unsigned short Klds[2][64 * 256];  // [buf][key][d]   2x32KB
  __shared__ unsigned short Vlds[2][256 * 64];  // [buf][d][key]   2x32KB
  __shared__ unsigned short Pl[4][16 * 64];     // per-wave P tile 8KB

  const int tid = threadIdx.x;
  const int lane = tid & 31;
  const int wave = tid >> 5;
  const int l16 = lane & 15;
  const int half = lane >> 4;

  const int b = blockIdx.y;
  const int q0 = blockIdx.x * 64 + wave * 16;
  const int qbase = b * 4096 + q0;

  // Resident Q A-tiles (already scaled by (1/sqrt(D))*log2e).
  v16bf qreg[8];
#pragma unroll
  for (int dblk = 0; dblk < 8; ++dblk)
    qreg[dblk] = load16(&Qw[(qbase + l16) * 256 + dblk * 32 + half * 8]);

  v8f acc[16];
#pragma unroll
  for (int nt = 0; nt < 16; ++nt) acc[nt] = vzero8();

  float mrun[8], lrun[8];
#pragma unroll
  for (int r = 0; r < 8; ++r) {
    mrun[r] = -INFINITY;
    lrun[r] = 0.0f;
  }

  const unsigned short* kbat = &Kw[(size_t)b * 4096 * 256];
  const unsigned short* vbat = &Vtw[(size_t)b * 256 * 4096];

  // Stage key-block kb into buffer buf (async, 16B per lane-op).
  auto stage = [&](int kb, int buf) {
    const unsigned short* ks = kbat + (size_t)kb * 64 * 256;  // 64x256 rows
    for (int i = tid; i < 2048; i += 128)
      async_copy_b128(&Klds[buf][i * 8], ks + i * 8);
    const unsigned short* vs = vbat + kb * 64;  // 256 rows of 64 keys
    for (int i = tid; i < 2048; i += 128) {
      const int d = i >> 3, ch = i & 7;
      async_copy_b128(&Vlds[buf][d * 64 + ch * 8],
                      vs + (size_t)d * 4096 + ch * 8);
    }
  };

  stage(0, 0);
  wait_async0();
  __syncthreads();

  for (int kb = 0; kb < 64; ++kb) {
    const int buf = kb & 1;
    // Prefetch next block into the other buffer (its readers finished at the
    // barrier that ended iteration kb-1).
    if (kb + 1 < 64) stage(kb + 1, buf ^ 1);

    // S = Q K^T : 16 queries x 64 keys (four 16x16 C tiles), base-2 domain.
    v8f s[4];
#pragma unroll
    for (int t = 0; t < 4; ++t) s[t] = vzero8();
#pragma unroll
    for (int dblk = 0; dblk < 8; ++dblk) {
#pragma unroll
      for (int t = 0; t < 4; ++t) {
        const v16bf bt =
            load16(&Klds[buf][(t * 16 + l16) * 256 + dblk * 32 + half * 8]);
        s[t] = wmma_bf16(qreg[dblk], bt, s[t]);
      }
    }

    // Online softmax (exp2); row = r + 8*half, cols across 16-lane half.
    float corr[8];
#pragma unroll
    for (int r = 0; r < 8; ++r) {
      const float a0 = s[0][r], a1 = s[1][r], a2 = s[2][r], a3 = s[3][r];
      float mx = fmaxf(fmaxf(a0, a1), fmaxf(a2, a3));
      mx = fmaxf(mx, __shfl_xor(mx, 1, 32));
      mx = fmaxf(mx, __shfl_xor(mx, 2, 32));
      mx = fmaxf(mx, __shfl_xor(mx, 4, 32));
      mx = fmaxf(mx, __shfl_xor(mx, 8, 32));
      const float mnew = fmaxf(mrun[r], mx);
      const float c = __builtin_amdgcn_exp2f(mrun[r] - mnew);
      const float p0 = __builtin_amdgcn_exp2f(a0 - mnew);
      const float p1 = __builtin_amdgcn_exp2f(a1 - mnew);
      const float p2 = __builtin_amdgcn_exp2f(a2 - mnew);
      const float p3 = __builtin_amdgcn_exp2f(a3 - mnew);
      float rs = (p0 + p1) + (p2 + p3);
      rs += __shfl_xor(rs, 1, 32);
      rs += __shfl_xor(rs, 2, 32);
      rs += __shfl_xor(rs, 4, 32);
      rs += __shfl_xor(rs, 8, 32);
      lrun[r] = lrun[r] * c + rs;
      mrun[r] = mnew;
      corr[r] = c;
      unsigned short* prow = &Pl[wave][(r + 8 * half) * 64 + l16];
      prow[0] = f2bf(p0);
      prow[16] = f2bf(p1);
      prow[32] = f2bf(p2);
      prow[48] = f2bf(p3);
    }

    // Rescale running output accumulators.
#pragma unroll
    for (int nt = 0; nt < 16; ++nt)
#pragma unroll
      for (int r = 0; r < 8; ++r) acc[nt][r] *= corr[r];

    // Intra-wave LDS RAW fence for the P tile round-trip.
    asm volatile("s_wait_dscnt 0" ::: "memory");

    const v16bf pa0 = load16(&Pl[wave][l16 * 64 + half * 8]);
    const v16bf pa1 = load16(&Pl[wave][l16 * 64 + 32 + half * 8]);
#pragma unroll
    for (int nt = 0; nt < 16; ++nt) {
      const v16bf b0 = load16(&Vlds[buf][(nt * 16 + l16) * 64 + half * 8]);
      acc[nt] = wmma_bf16(pa0, b0, acc[nt]);
      const v16bf b1 = load16(&Vlds[buf][(nt * 16 + l16) * 64 + 32 + half * 8]);
      acc[nt] = wmma_bf16(pa1, b1, acc[nt]);
    }

    // Make prefetched buffer visible / protect it from early overwrite.
    wait_async0();
    __syncthreads();
  }

  // Normalize and store fp32 output.
  float inv[8];
#pragma unroll
  for (int r = 0; r < 8; ++r) inv[r] = 1.0f / lrun[r];
#pragma unroll
  for (int nt = 0; nt < 16; ++nt)
#pragma unroll
    for (int r = 0; r < 8; ++r)
      out[(b * 4096 + q0 + r + 8 * half) * 256 + nt * 16 + l16] =
          acc[nt][r] * inv[r];
}

// ---------------------------------------------------------------------------
// Launch.  d_in order: x, Wk, bk, Wq, bq, Wv, bv.
// Workspace: bf16 Q | K | V^T, each 16384*256 elements (24 MB total).
// ---------------------------------------------------------------------------
extern "C" void kernel_launch(void* const* d_in, const int* in_sizes, int n_in,
                              void* d_out, int out_size, void* d_ws,
                              size_t ws_size, hipStream_t stream) {
  const float* x = (const float*)d_in[0];
  const float* Wk = (const float*)d_in[1];
  const float* bk = (const float*)d_in[2];
  const float* Wq = (const float*)d_in[3];
  const float* bq = (const float*)d_in[4];
  const float* Wv = (const float*)d_in[5];
  const float* bv = (const float*)d_in[6];
  float* out = (float*)d_out;

  unsigned short* Qw = (unsigned short*)d_ws;
  unsigned short* Kw = Qw + (size_t)16384 * 256;
  unsigned short* Vtw = Kw + (size_t)16384 * 256;

  qkv_proj_kernel<<<dim3(256, 3), 128, 0, stream>>>(x, Wq, bq, Wk, bk, Wv, bv,
                                                    Qw, Kw, Vtw);
  flash_attn_kernel<<<dim3(64, 4), 128, 0, stream>>>(Qw, Kw, Vtw, out);
}
